// ModulatedConv2d_55765855371747
// MI455X (gfx1250) — matrix operand order repaired and checked by reading, not compiled, not run on hardware
//
#include <hip/hip_runtime.h>
#include <hip/hip_bf16.h>
#include <stdint.h>

// ---------------- problem constants ----------------
#define NB   8      // batch
#define CIN  512    // in channels
#define HH   64
#define WW   64
#define OC   512    // out channels
#define KK9  9      // 3x3
#define KTOT (CIN*KK9)     // 4608
#define HWP  (HH*WW)       // 4096
#define SCH  512           // style_c

#define S_SCALE 0.044194173824159216f   // 1/sqrt(512)
#define W_SCALE 0.014731391274719739f   // 1/sqrt(4608)
#define EPSV    1e-8f

// ---------------- GEMM tiling ----------------
#define BM 128
#define BN 128
#define BKC 32              // input channels per super-chunk
#define BK  (BKC*KK9)       // 288 K per super-chunk
#define A_STRIDE 296        // bf16 elems per LDS A row (288 -> 296: 16B-aligned, 37*16B stride)
#define BT_STRIDE 148       // u32 (K-pairs) per LDS B row (144 -> 148: 16B-aligned, 37*16B stride)
#define A_BYTES  (BM*A_STRIDE*2)            // 75776
#define B_BYTES  (BN*BT_STRIDE*4)           // 75776
#define SMEM_BYTES (A_BYTES + B_BYTES)      // 151552

typedef __attribute__((ext_vector_type(16))) __bf16 v16bf;
typedef __attribute__((ext_vector_type(8)))  float  v8f;

union Frag {
    uint4    u4[2];
    uint32_t u[8];
    v16bf    v;
};

__device__ __forceinline__ uint32_t bf16_rne(float f) {
    uint32_t u = __float_as_uint(f);
    u = u + 0x7FFFu + ((u >> 16) & 1u);     // round-to-nearest-even
    return u >> 16;
}

// CDNA5 async global->LDS copy (16B per lane), tracked by ASYNCcnt
__device__ __forceinline__ void async_copy_b128(uint32_t lds_off, const void* g) {
    asm volatile("global_load_async_to_lds_b128 %0, %1, off"
                 :: "v"(lds_off), "v"((uint64_t)(uintptr_t)g)
                 : "memory");
}
__device__ __forceinline__ void wait_async0() {
    asm volatile("s_wait_asynccnt 0" ::: "memory");
}

// ---------------- kernel 1: style modulation s[n][c] ----------------
__global__ void __launch_bounds__(256)
k_style(const float* __restrict__ style, const float* __restrict__ style_w,
        const float* __restrict__ style_b, float* __restrict__ s) {
    int idx = blockIdx.x * 256 + threadIdx.x;          // 0..4095
    int n = idx >> 9, c = idx & 511;
    const float* st = style + n * SCH;
    const float* sw = style_w + c * SCH;
    float acc = 0.f;
    for (int j = 0; j < SCH; ++j) acc += st[j] * sw[j];
    s[idx] = acc * S_SCALE + style_b[c];
}

// ---------------- kernel 2: demod[n][o] (factorized) ----------------
__global__ void __launch_bounds__(256)
k_demod(const float* __restrict__ weight, const float* __restrict__ s,
        float* __restrict__ demod) {
    int idx = blockIdx.x * 256 + threadIdx.x;          // 0..4095
    int n = idx >> 9, o = idx & 511;
    const float* sv = s + n * CIN;
    float acc = 0.f;
    for (int i = 0; i < CIN; ++i) {
        float si = sv[i]; si *= si;
        const float* wp = weight + (size_t)(o * CIN + i) * KK9;
        float wq = 0.f;
        #pragma unroll
        for (int r = 0; r < KK9; ++r) { float w = wp[r]; wq += w * w; }
        acc += si * wq;
    }
    demod[idx] = rsqrtf(acc * (W_SCALE * W_SCALE) + EPSV);
}

// ---------------- kernel 3: modulated bf16 weights wmodb[n][o][k] ----------------
__global__ void __launch_bounds__(256)
k_modw(const float* __restrict__ weight, const float* __restrict__ s,
       const float* __restrict__ demod, uint32_t* __restrict__ wmodb2) {
    // one u32 = two consecutive bf16 K elements. total pairs = 8*512*4608/2
    int64_t pidx = (int64_t)blockIdx.x * 256 + threadIdx.x;
    int64_t e = pidx * 2;
    int n  = (int)(e / ((int64_t)OC * KTOT));
    int rem = (int)(e - (int64_t)n * OC * KTOT);
    int o  = rem / KTOT;
    int kk = rem - o * KTOT;
    float d = demod[n * OC + o] * W_SCALE;
    uint32_t packed = 0;
    #pragma unroll
    for (int h = 0; h < 2; ++h) {
        int k = kk + h;
        int i = k / KK9;
        float f = weight[(size_t)o * KTOT + k] * s[n * CIN + i] * d;
        packed |= bf16_rne(f) << (16 * h);
    }
    wmodb2[pidx] = packed;
}

// ---------------- kernel 4: per-sample implicit-GEMM conv via WMMA ----------------
__global__ void __launch_bounds__(256)
k_conv(const float* __restrict__ x, const uint16_t* __restrict__ wmodb,
       float* __restrict__ out) {
    extern __shared__ char smem[];
    uint16_t* aTile = (uint16_t*)smem;                 // [BM][A_STRIDE] bf16 bits
    uint32_t* bTile = (uint32_t*)(smem + A_BYTES);     // [BN][BT_STRIDE] packed K-pairs (transposed)
    const uint32_t aBase = (uint32_t)(uintptr_t)aTile; // LDS byte offset of A tile

    const int tid  = threadIdx.x;
    const int lane = tid & 31;
    const int wv   = tid >> 5;     // 0..7
    const int wm   = wv >> 2;      // 0..1 -> M offset 64*wm
    const int wn   = wv & 3;       // 0..3 -> N offset 32*wn
    const int P0   = blockIdx.x * BN;      // pixel tile
    const int O0   = blockIdx.y * BM;      // out-channel tile
    const int n    = blockIdx.z;

    const uint16_t* wrow = wmodb + (size_t)(n * OC + O0) * KTOT;
    const float*    xs   = x + (size_t)n * CIN * HWP;

    v8f acc[4][2];
    #pragma unroll
    for (int mi = 0; mi < 4; ++mi)
        #pragma unroll
        for (int ni = 0; ni < 2; ++ni)
            acc[mi][ni] = (v8f){0.f,0.f,0.f,0.f,0.f,0.f,0.f,0.f};

    const int hsel = (lane & 16) ? 1 : 0;  // half-wave selector
    const int lm   = lane & 15;

    for (int sc = 0; sc < CIN / BKC; ++sc) {
        const int ci0 = sc * BKC;
        // ---- stage A: contiguous async copy of BM x 288 bf16 (16B per lane) ----
        #pragma unroll 2
        for (int it = 0; it < (BM * (BK / 8)) / 256; ++it) {   // 18 iters
            int seg = tid + it * 256;
            int row = seg / (BK / 8);
            int c16 = seg - row * (BK / 8);
            async_copy_b128(aBase + (uint32_t)(row * A_STRIDE + c16 * 8) * 2,
                            wrow + (size_t)row * KTOT + ci0 * KK9 + c16 * 8);
        }
        // ---- stage B: on-the-fly im2col, packed bf16 K-pairs, transposed [p][kp] ----
        for (int it = 0; it < (BK / 2 * BN) / 256; ++it) {     // 72 iters
            int pidx = tid + it * 256;
            int kp = pidx >> 7;            // 0..143
            int p  = pidx & 127;
            int pg = P0 + p;
            int yy = pg >> 6, xx = pg & 63;
            uint32_t packed = 0;
            #pragma unroll
            for (int h = 0; h < 2; ++h) {
                int k  = kp * 2 + h;
                int il = k / KK9;
                int r  = k - il * KK9;
                int ky = r / 3;
                int kx = r - ky * 3;
                int sy = yy + ky - 1, sx = xx + kx - 1;
                float f = 0.f;
                if ((unsigned)sy < (unsigned)HH && (unsigned)sx < (unsigned)WW)
                    f = xs[(size_t)((ci0 + il) * HH + sy) * WW + sx];
                packed |= bf16_rne(f) << (16 * h);
            }
            bTile[p * BT_STRIDE + kp] = packed;
        }
        wait_async0();
        __syncthreads();
        // ---- compute: 9 WMMA K-steps of 32 ----
        for (int ks = 0; ks < KK9; ++ks) {
            Frag af[4];
            #pragma unroll
            for (int mi = 0; mi < 4; ++mi) {
                int m = wm * 64 + mi * 16 + lm;
                const char* rp = (const char*)(aTile + m * A_STRIDE) + ks * 64 + hsel * 16;
                af[mi].u4[0] = *(const uint4*)rp;
                af[mi].u4[1] = *(const uint4*)(rp + 32);
            }
            Frag bfr[2];
            #pragma unroll
            for (int ni = 0; ni < 2; ++ni) {
                int col = wn * 32 + ni * 16 + lm;
                const char* bp = (const char*)(bTile + col * BT_STRIDE) + ks * 64 + hsel * 16;
                bfr[ni].u4[0] = *(const uint4*)bp;
                bfr[ni].u4[1] = *(const uint4*)(bp + 32);
            }
            #pragma unroll
            for (int mi = 0; mi < 4; ++mi)
                #pragma unroll
                for (int ni = 0; ni < 2; ++ni)
                    acc[mi][ni] = __builtin_amdgcn_wmma_f32_16x16x32_bf16(
                        false, af[mi].v, false, bfr[ni].v,
                        (short)0, acc[mi][ni], false, false);
        }
        __syncthreads();
    }

    // ---- epilogue: C/D layout -> global (M = vgpr + 8*(lane>=16), N = lane&15) ----
    float* op = out + (size_t)n * OC * HWP;
    const int mh = hsel * 8;
    #pragma unroll
    for (int mi = 0; mi < 4; ++mi) {
        #pragma unroll
        for (int ni = 0; ni < 2; ++ni) {
            int pcol  = P0 + wn * 32 + ni * 16 + lm;
            int obase = O0 + wm * 64 + mi * 16 + mh;
            #pragma unroll
            for (int v = 0; v < 8; ++v)
                op[(size_t)(obase + v) * HWP + pcol] = acc[mi][ni][v];
        }
    }
}

// ---------------- launcher ----------------
extern "C" void kernel_launch(void* const* d_in, const int* in_sizes, int n_in,
                              void* d_out, int out_size, void* d_ws, size_t ws_size,
                              hipStream_t stream) {
    const float* x       = (const float*)d_in[0];
    const float* style   = (const float*)d_in[1];
    const float* weight  = (const float*)d_in[2];
    const float* style_w = (const float*)d_in[3];
    const float* style_b = (const float*)d_in[4];
    float* out = (float*)d_out;

    float*    s     = (float*)d_ws;                             // 8*512 f32
    float*    demod = (float*)((char*)d_ws + 16384);            // 8*512 f32
    uint16_t* wmodb = (uint16_t*)((char*)d_ws + 32768);         // 8*512*4608 bf16

    k_style<<<16, 256, 0, stream>>>(style, style_w, style_b, s);
    k_demod<<<16, 256, 0, stream>>>(weight, s, demod);

    int64_t pairs = (int64_t)NB * OC * KTOT / 2;                // 9,437,184
    k_modw<<<(int)(pairs / 256), 256, 0, stream>>>(weight, s, demod,
                                                   (uint32_t*)wmodb);

    dim3 grid(HWP / BN, OC / BM, NB);                           // (32, 4, 8)
    k_conv<<<grid, 256, SMEM_BYTES, stream>>>(x, wmodb, out);
}
